// BerryPhaseInterferometer_56126632624432
// MI455X (gfx1250) — compile-verified, matrix operand order checked by reading, not run on hardware
//
#include <hip/hip_runtime.h>
#include <hip/hip_bf16.h>
#include <math.h>

typedef __attribute__((ext_vector_type(16))) _Float16 v16h;
typedef __attribute__((ext_vector_type(8)))  float    v8f;

#define TILE_B 16
#define DIMF   256
#define HALF   128
#define NBLK   256   // 4096 / 16

__device__ __forceinline__ float gelu_exact(float x) {
    // exact GELU: 0.5*x*(1+erf(x/sqrt(2)))  (reference uses approximate=False)
    return 0.5f * x * (1.0f + erff(x * 0.70710678118654752f));
}

// Process one modality for a 16-row batch tile.
// Leaves the reversible-coupled state (16 x 256, f32) in `st`.
__device__ __forceinline__ void modality_tile(
    const float* __restrict__ x, int in_f,
    const float* __restrict__ pw, const float* __restrict__ pb,
    const float* __restrict__ hh, const float* __restrict__ dl,
    const float* __restrict__ Fw, const float* __restrict__ Fb,
    const float* __restrict__ Gw, const float* __restrict__ Gb,
    float* st, _Float16* aX, _Float16* aY, int tid)
{
    // ---- Phase 1: fractal expand: knot = tanh(base*(h+d) - base*(h-d)) ----
    {
        int c = tid;                       // one column per thread, 16 rows each
        float w0 = pw[c*in_f + 0], w1 = 0.f, w2 = 0.f;
        if (in_f > 1) { w1 = pw[c*in_f + 1]; w2 = pw[c*in_f + 2]; }
        float bias = pb[c], h = hh[c], d = dl[c];
        float hp = h + d, hm = h - d;
        #pragma unroll
        for (int r = 0; r < TILE_B; ++r) {
            float base = x[r*in_f + 0]*w0 + bias;
            if (in_f > 1) base += x[r*in_f + 1]*w1 + x[r*in_f + 2]*w2;
            float knot = tanhf(base*hp - base*hm);
            st[r*DIMF + c] = knot;
            if (c >= HALF) aX[r*HALF + (c - HALF)] = (_Float16)knot;  // x2 in f16
        }
    }
    __syncthreads();

    const int wave = tid >> 5, lane = tid & 31;
    const int hi = lane >> 4, lm = lane & 15;
    const int n = wave*16 + lm;            // output column owned by this lane

    // ---- Phase 2: y1 = x1 + gelu(x2 @ F^T + Fb) ----
    {
        v8f acc = {};
        #pragma unroll
        for (int kc = 0; kc < 4; ++kc) {
            // 16-bit A 16x32 layout: lane<16 -> K {0..7, 16..23}, lane>=16 -> +8
            const _Float16* ap = aX + lm*HALF + kc*32 + 8*hi;
            const float*    bp = Fw + n*HALF  + kc*32 + 8*hi;   // B[k][n] = Fw[n][k]
            v16h a, b;
            #pragma unroll
            for (int j = 0; j < 8; ++j) {
                a[j]   = ap[j];          a[8+j] = ap[16+j];
                b[j]   = (_Float16)bp[j];
                b[8+j] = (_Float16)bp[16+j];
            }
            acc = __builtin_amdgcn_wmma_f32_16x16x32_f16(false, a, false, b,
                                                         (short)0, acc, false, false);
        }
        float fb = Fb[n];
        #pragma unroll
        for (int i = 0; i < 8; ++i) {
            int m = i + 8*hi;                          // C/D layout: VGPR i -> M
            float y1 = st[m*DIMF + n] + gelu_exact(acc[i] + fb);
            st[m*DIMF + n] = y1;                       // overwrite x1 half
            aY[m*HALF + n] = (_Float16)y1;             // A-operand for G-GEMM
        }
    }
    __syncthreads();

    // ---- Phase 3: y2 = x2 + gelu(y1 @ G^T + Gb) ----
    {
        v8f acc = {};
        #pragma unroll
        for (int kc = 0; kc < 4; ++kc) {
            const _Float16* ap = aY + lm*HALF + kc*32 + 8*hi;
            const float*    bp = Gw + n*HALF  + kc*32 + 8*hi;
            v16h a, b;
            #pragma unroll
            for (int j = 0; j < 8; ++j) {
                a[j]   = ap[j];          a[8+j] = ap[16+j];
                b[j]   = (_Float16)bp[j];
                b[8+j] = (_Float16)bp[16+j];
            }
            acc = __builtin_amdgcn_wmma_f32_16x16x32_f16(false, a, false, b,
                                                         (short)0, acc, false, false);
        }
        float gb = Gb[n];
        #pragma unroll
        for (int i = 0; i < 8; ++i) {
            int m = i + 8*hi;
            float y2 = st[m*DIMF + HALF + n] + gelu_exact(acc[i] + gb);
            st[m*DIMF + HALF + n] = y2;                // overwrite x2 half
        }
    }
    __syncthreads();
}

__global__ __launch_bounds__(256)
void berry_main(const float* __restrict__ vis,  const float* __restrict__ txt,
                const float* __restrict__ vpw,  const float* __restrict__ vpb,
                const float* __restrict__ vh,   const float* __restrict__ vdl,
                const float* __restrict__ tpw,  const float* __restrict__ tpb,
                const float* __restrict__ th,   const float* __restrict__ tdl,
                const float* __restrict__ Fw,   const float* __restrict__ Fb,
                const float* __restrict__ Gw,   const float* __restrict__ Gb,
                float* __restrict__ out, float* __restrict__ eta_part)
{
    __shared__ float    st_v[TILE_B * DIMF];   // 16 KB
    __shared__ float    st_t[TILE_B * DIMF];   // 16 KB
    __shared__ _Float16 aX[TILE_B * HALF];     // 4 KB
    __shared__ _Float16 aY[TILE_B * HALF];     // 4 KB
    __shared__ float    red_re[256], red_im[256];
    __shared__ float    etabuf[TILE_B];

    const int tid = threadIdx.x;
    const int b0  = blockIdx.x * TILE_B;

    modality_tile(vis + (size_t)b0*3, 3, vpw, vpb, vh, vdl,
                  Fw, Fb, Gw, Gb, st_v, aX, aY, tid);
    __syncthreads();   // aX/aY reuse boundary
    modality_tile(txt + (size_t)b0*1, 1, tpw, tpb, th, tdl,
                  Fw, Fb, Gw, Gb, st_t, aX, aY, tid);
    __syncthreads();

    // ---- inner_b = sum_k psi_v[k] * conj(psi_t[k]) ----
    {
        int r = tid >> 4, seg = tid & 15;
        float re = 0.f, im = 0.f;
        #pragma unroll
        for (int q = 0; q < 8; ++q) {
            int k = seg*8 + q;
            float a = st_v[r*DIMF + 2*k],     b = st_v[r*DIMF + 2*k + 1];
            float c = st_t[r*DIMF + 2*k],     d = st_t[r*DIMF + 2*k + 1];
            re += a*c + b*d;            // Re[(a+bi)(c-di)]
            im += b*c - a*d;            // Im[(a+bi)(c-di)]
        }
        red_re[tid] = re; red_im[tid] = im;
    }
    __syncthreads();

    if (tid < TILE_B) {
        float R = 0.f, I = 0.f;
        #pragma unroll
        for (int s = 0; s < 16; ++s) { R += red_re[tid*16 + s]; I += red_im[tid*16 + s]; }
        out[b0 + tid] = atan2f(I, R);                       // berry phase
        // det(S/EPS + I) == 1 + inner/EPS  (rank-1 update identity)
        etabuf[tid] = atan2f(I * 1.0e6f, 1.0f + R * 1.0e6f);
    }
    __syncthreads();
    if (tid == 0) {
        float s = 0.f;
        #pragma unroll
        for (int i = 0; i < TILE_B; ++i) s += etabuf[i];
        eta_part[blockIdx.x] = s;                            // deterministic partials
    }
}

__global__ __launch_bounds__(256)
void berry_finish(const float* __restrict__ eta_part, float* __restrict__ out)
{
    __shared__ float s[256];
    int t = threadIdx.x;
    s[t] = eta_part[t];
    __syncthreads();
    #pragma unroll
    for (int off = 128; off > 0; off >>= 1) {
        if (t < off) s[t] += s[t + off];
        __syncthreads();
    }
    if (t == 0) out[4096] = s[0] * (1.0f / (4096.0f * 3.14159265358979323846f));
}

extern "C" void kernel_launch(void* const* d_in, const int* in_sizes, int n_in,
                              void* d_out, int out_size, void* d_ws, size_t ws_size,
                              hipStream_t stream) {
    const float* vis = (const float*)d_in[0];   // (4096,3)
    const float* txt = (const float*)d_in[1];   // (4096,1)
    const float* vpw = (const float*)d_in[2];   // (256,3)
    const float* vpb = (const float*)d_in[3];   // (256,)
    const float* vh  = (const float*)d_in[4];   // (1,256)
    const float* vdl = (const float*)d_in[5];   // (1,256)
    const float* tpw = (const float*)d_in[6];   // (256,1)
    const float* tpb = (const float*)d_in[7];   // (256,)
    const float* th  = (const float*)d_in[8];   // (1,256)
    const float* tdl = (const float*)d_in[9];   // (1,256)
    const float* Fw  = (const float*)d_in[10];  // (128,128)
    const float* Fb  = (const float*)d_in[11];  // (128,)
    const float* Gw  = (const float*)d_in[12];  // (128,128)
    const float* Gb  = (const float*)d_in[13];  // (128,)

    float* out      = (float*)d_out;            // 4096 phases + 1 eta
    float* eta_part = (float*)d_ws;             // 256 floats

    berry_main<<<NBLK, 256, 0, stream>>>(vis, txt, vpw, vpb, vh, vdl,
                                         tpw, tpb, th, tdl,
                                         Fw, Fb, Gw, Gb, out, eta_part);
    berry_finish<<<1, 256, 0, stream>>>(eta_part, out);
}
